// Categorical_78494822302255
// MI455X (gfx1250) — matrix operand order repaired and checked by reading, not compiled, not run on hardware
//
#include <hip/hip_runtime.h>
#include <cstdint>
#include <cstddef>

// ---------------- problem constants ----------------
#define N_ROWS   16384
#define DIN      1024
#define DOUT     512
#define NEXP     8
#define MAX_TILES 1040            // >= N/16 + E, rounded up
// LDS staging: 16 rows x 512 K of bf16, hi & lo planes
#define KCHUNK   512
#define ROWSTRIDE 1040            // bytes: 512*2 + 16B pad -> bank advance of 4
#define LDS_HALF (16 * ROWSTRIDE) // 16640 B
#define LDS_TOTAL (2 * LDS_HALF)  // 33280 B

typedef __bf16 v16bf __attribute__((ext_vector_type(16)));
typedef float  v8f   __attribute__((ext_vector_type(8)));

union Frag16 { uint4 q[2]; v16bf v; };   // 32 bytes: 16 bf16

__device__ __forceinline__ uint16_t bf16_rne(float f) {
    uint32_t u = __float_as_uint(f);
    return (uint16_t)((u + 0x7FFFu + ((u >> 16) & 1u)) >> 16);
}
__device__ __forceinline__ float bf16f(uint16_t h) {
    return __uint_as_float(((uint32_t)h) << 16);
}

// ---------------- routing kernels ----------------
__global__ void k_init(int* __restrict__ row_map, uint32_t* __restrict__ counts,
                       uint32_t* __restrict__ cursors) {
    int i = blockIdx.x * blockDim.x + threadIdx.x;
    if (i < MAX_TILES * 16) row_map[i] = -1;
    if (i < NEXP) { counts[i] = 0u; cursors[i] = 0u; }
}

__global__ void k_count(const int* __restrict__ index, uint32_t* __restrict__ counts) {
    int i = blockIdx.x * blockDim.x + threadIdx.x;
    if (i < N_ROWS) atomicAdd(&counts[index[i]], 1u);
}

__global__ void k_scan(const uint32_t* __restrict__ counts, uint32_t* __restrict__ offsets,
                       int* __restrict__ tile_expert, uint32_t* __restrict__ meta) {
    if (threadIdx.x != 0 || blockIdx.x != 0) return;
    uint32_t off = 0, tid = 0;
    for (int e = 0; e < NEXP; ++e) {
        offsets[e] = off;
        uint32_t nt = (counts[e] + 15u) >> 4;
        for (uint32_t t = 0; t < nt; ++t) tile_expert[tid++] = e;
        off += nt * 16u;
    }
    meta[0] = tid;  // number of active 16-row tiles
}

__global__ void k_scatter(const int* __restrict__ index, const uint32_t* __restrict__ offsets,
                          uint32_t* __restrict__ cursors, int* __restrict__ row_map) {
    int i = blockIdx.x * blockDim.x + threadIdx.x;
    if (i >= N_ROWS) return;
    int e = index[i];
    uint32_t p = atomicAdd(&cursors[e], 1u);
    row_map[offsets[e] + p] = i;
}

// ---------------- W -> bf16 hi/lo, kept in [e][n][k] layout (k fastest) ----------------
__global__ void k_convert_w(const float* __restrict__ W,
                            uint16_t* __restrict__ wt_hi, uint16_t* __restrict__ wt_lo) {
    size_t flat = (size_t)blockIdx.x * blockDim.x + threadIdx.x; // [e][n][k], k fastest
    if (flat >= (size_t)NEXP * DIN * DOUT) return;
    int k = (int)(flat & (DIN - 1));
    int n = (int)((flat >> 10) & (DOUT - 1));
    int e = (int)(flat >> 19);
    float v = W[((size_t)e * DOUT + n) * DIN + k];
    uint16_t hi = bf16_rne(v);
    uint16_t lo = bf16_rne(v - bf16f(hi));
    wt_hi[flat] = hi;
    wt_lo[flat] = lo;
}

// ---------------- grouped GEMM: 16 rows x 512 cols per block ----------------
__global__ __launch_bounds__(256) void k_gemm(
    const float* __restrict__ x,
    const uint16_t* __restrict__ wt_hi, const uint16_t* __restrict__ wt_lo,
    const float* __restrict__ bias,
    const int* __restrict__ row_map, const int* __restrict__ tile_expert,
    const uint32_t* __restrict__ meta, float* __restrict__ out)
{
    extern __shared__ unsigned char smem[];
    unsigned char* lds_hi = smem;
    unsigned char* lds_lo = smem + LDS_HALF;

    const int tile = blockIdx.x;
    if (tile >= (int)meta[0]) return;          // uniform early exit
    const int e = tile_expert[tile];
    const int t = threadIdx.x;
    const int lane = t & 31;
    const int wv   = t >> 5;
    const int nbase = wv * 64;                 // this wave's 64 output columns
    const int m   = lane & 15;                 // A row within tile
    const int sel = (lane >> 4) * 8;           // K sub-select per 16-bit A layout

    v8f zero = {0.f, 0.f, 0.f, 0.f, 0.f, 0.f, 0.f, 0.f};
    v8f c[4];
#pragma unroll
    for (int j = 0; j < 4; ++j) c[j] = zero;

    const size_t wexp = (size_t)e * DIN * DOUT;

    for (int ks = 0; ks < DIN; ks += KCHUNK) {
        __syncthreads();   // all waves done reading previous stage
        // ---- stage 16 x 512 fp32 -> bf16 hi/lo into LDS (coalesced b128 loads) ----
#pragma unroll
        for (int it = 0; it < 8; ++it) {
            int lin = t + it * 256;            // float4 index over 16 x 128
            int row = lin >> 7;
            int k4  = lin & 127;
            int grow = row_map[tile * 16 + row];
            float4 v = make_float4(0.f, 0.f, 0.f, 0.f);
            if (grow >= 0)
                v = *(const float4*)(x + (size_t)grow * DIN + ks + k4 * 4);
            uint16_t h0 = bf16_rne(v.x), h1 = bf16_rne(v.y), h2 = bf16_rne(v.z), h3 = bf16_rne(v.w);
            uint16_t l0 = bf16_rne(v.x - bf16f(h0)), l1 = bf16_rne(v.y - bf16f(h1));
            uint16_t l2 = bf16_rne(v.z - bf16f(h2)), l3 = bf16_rne(v.w - bf16f(h3));
            int boff = row * ROWSTRIDE + k4 * 8;
            *(uint2*)(lds_hi + boff) = make_uint2((uint32_t)h0 | ((uint32_t)h1 << 16),
                                                  (uint32_t)h2 | ((uint32_t)h3 << 16));
            *(uint2*)(lds_lo + boff) = make_uint2((uint32_t)l0 | ((uint32_t)l1 << 16),
                                                  (uint32_t)l2 | ((uint32_t)l3 << 16));
        }
        __syncthreads();

        // ---- compute over this 512-K chunk ----
        for (int kc = 0; kc < KCHUNK; kc += 32) {
            Frag16 ahi, alo;
            int aoff = m * ROWSTRIDE + (kc + sel) * 2;
            ahi.q[0] = *(const uint4*)(lds_hi + aoff);        // K kc..kc+7  (or +8..15)
            ahi.q[1] = *(const uint4*)(lds_hi + aoff + 32);   // K kc+16..23 (or +24..31)
            alo.q[0] = *(const uint4*)(lds_lo + aoff);
            alo.q[1] = *(const uint4*)(lds_lo + aoff + 32);

            // B: lane <-> N = nbase+(lane&15) ; K kc+sel..+7 and kc+16+sel..+7 per lane
            const size_t brow = wexp + (size_t)(nbase + m) * DIN + ks + kc + sel;
            const uint16_t* bh = wt_hi + brow;
            const uint16_t* bl = wt_lo + brow;
#pragma unroll
            for (int ct = 0; ct < 4; ++ct) {
                Frag16 bhi, blo;
                bhi.q[0] = *(const uint4*)(bh + ct * 16 * DIN);
                bhi.q[1] = *(const uint4*)(bh + ct * 16 * DIN + 16);
                blo.q[0] = *(const uint4*)(bl + ct * 16 * DIN);
                blo.q[1] = *(const uint4*)(bl + ct * 16 * DIN + 16);
                c[ct] = __builtin_amdgcn_wmma_f32_16x16x32_bf16(
                            false, ahi.v, false, bhi.v, (short)0, c[ct], false, false);
                c[ct] = __builtin_amdgcn_wmma_f32_16x16x32_bf16(
                            false, ahi.v, false, blo.v, (short)0, c[ct], false, false);
                c[ct] = __builtin_amdgcn_wmma_f32_16x16x32_bf16(
                            false, alo.v, false, bhi.v, (short)0, c[ct], false, false);
            }
        }
    }

    // ---- epilogue: bias + scatter to routed rows ----
    // C layout: VGPR j -> M = j (lanes 0-15) or j+8 (lanes 16-31); N = lane&15
    const int coln = lane & 15;
    const int rbase = tile * 16 + sel;   // sel==8 for upper half lanes
    float bv[4];
#pragma unroll
    for (int ct = 0; ct < 4; ++ct)
        bv[ct] = bias[e * DOUT + nbase + ct * 16 + coln];
#pragma unroll
    for (int j = 0; j < 8; ++j) {
        int grow = row_map[rbase + j];
        if (grow >= 0) {
#pragma unroll
            for (int ct = 0; ct < 4; ++ct)
                out[(size_t)grow * DOUT + nbase + ct * 16 + coln] = c[ct][j] + bv[ct];
        }
    }
}

// ---------------- host launcher ----------------
extern "C" void kernel_launch(void* const* d_in, const int* in_sizes, int n_in,
                              void* d_out, int out_size, void* d_ws, size_t ws_size,
                              hipStream_t stream) {
    (void)in_sizes; (void)n_in; (void)out_size; (void)ws_size;
    const float* x     = (const float*)d_in[0];
    const int*   index = (const int*)d_in[1];
    const float* W     = (const float*)d_in[2];
    const float* b     = (const float*)d_in[3];
    float*       out   = (float*)d_out;

    uint8_t* ws = (uint8_t*)d_ws;
    const size_t WTSZ = (size_t)NEXP * DIN * DOUT * sizeof(uint16_t);  // 8 MB
    uint16_t* wt_hi = (uint16_t*)ws;
    uint16_t* wt_lo = (uint16_t*)(ws + WTSZ);
    uint32_t* counts  = (uint32_t*)(ws + 2 * WTSZ);
    uint32_t* cursors = counts + 8;
    uint32_t* offsets = counts + 16;
    uint32_t* meta    = counts + 24;
    int* tile_expert  = (int*)(counts + 32);
    int* row_map      = tile_expert + MAX_TILES;

    // routing
    k_init<<<(MAX_TILES * 16 + 255) / 256, 256, 0, stream>>>(row_map, counts, cursors);
    k_count<<<(N_ROWS + 255) / 256, 256, 0, stream>>>(index, counts);
    k_scan<<<1, 32, 0, stream>>>(counts, offsets, tile_expert, meta);
    k_scatter<<<(N_ROWS + 255) / 256, 256, 0, stream>>>(index, offsets, cursors, row_map);

    // weight precision split, layout unchanged (L2-resident afterwards)
    k_convert_w<<<(int)(((size_t)NEXP * DIN * DOUT + 255) / 256), 256, 0, stream>>>(W, wt_hi, wt_lo);

    // grouped GEMM
    k_gemm<<<dim3(MAX_TILES), dim3(256), LDS_TOTAL, stream>>>(
        x, wt_hi, wt_lo, b, row_map, tile_expert, meta, out);
}
